// IndRNN_74174085202363
// MI455X (gfx1250) — compile-verified, matrix operand order checked by reading, not run on hardware
//
#include <hip/hip_runtime.h>

// ---------------------------------------------------------------------------
// IndRNN (2 layers) for MI455X / gfx1250, wave32.
//   layer:  u = x @ W^T + b  (GEMM, split-bf16 WMMA)
//           h_t = relu(u_t + w_hh * h_{t-1})  (serial scan over T)
// ---------------------------------------------------------------------------

#define T_SEQ   2048
#define BATCH   32
#define DDIM    512                       // D_IN == D_HID == 512
#define M_TOTAL (T_SEQ * BATCH)           // 65536 GEMM rows
#define M_TILE  64
#define BLOCK_THREADS 256

typedef __attribute__((ext_vector_type(16))) __bf16 v16bf;
typedef __attribute__((ext_vector_type(8)))  __bf16 v8bf;
typedef __attribute__((ext_vector_type(4)))  __bf16 v4bf;
typedef __attribute__((ext_vector_type(8)))  float  v8f;

// ------------------------- split f32 -> (hi, lo) bf16 ----------------------
__device__ __forceinline__ void split_one(float f, __bf16& hi, __bf16& lo) {
    unsigned u  = __builtin_bit_cast(unsigned, f);
    float    hf = __builtin_bit_cast(float, u & 0xFFFF0000u);  // exact in bf16
    hi = (__bf16)hf;
    lo = (__bf16)(f - hf);   // residual, |lo| < ulp(hi)
}

// Convert n4*4 floats into hi/lo bf16 planes (bandwidth-bound pass).
__global__ __launch_bounds__(256)
void split_kernel(const float* __restrict__ src,
                  __bf16* __restrict__ hi, __bf16* __restrict__ lo, int n4) {
    int i = blockIdx.x * blockDim.x + threadIdx.x;
    if (i >= n4) return;
    float4 v = ((const float4*)src)[i];
    v4bf h, l;
    __bf16 a, b;
    split_one(v.x, a, b); h[0] = a; l[0] = b;
    split_one(v.y, a, b); h[1] = a; l[1] = b;
    split_one(v.z, a, b); h[2] = a; l[2] = b;
    split_one(v.w, a, b); h[3] = a; l[3] = b;
    ((v4bf*)hi)[i] = h;
    ((v4bf*)lo)[i] = l;
}

// ------------------------- fragment load (16-bit A/B) ----------------------
// Per ISA 7.12.2: lane half selects K base (0 or 8); frag = 8 bf16 at k0 and
// 8 bf16 at k0+16 -> two 16-byte loads, both rows are K-contiguous in memory.
__device__ __forceinline__ v16bf load_frag(const __bf16* __restrict__ row, int k0) {
    v8bf a = *(const v8bf*)(row + k0);
    v8bf b = *(const v8bf*)(row + k0 + 16);
    return __builtin_shufflevector(a, b, 0,1,2,3,4,5,6,7,8,9,10,11,12,13,14,15);
}

// ------------------------- GEMM: U = A(M,K) * W(N,K)^T + bias --------------
// Block: 256 thr = 8 waves. Block tile: M=64, N=512 (full width -> A read 1x).
// Wave w owns N in [64w, 64w+64): 4 N-subtiles x 4 M-subtiles of 16x16.
// Split-bf16: D += Ahi*Bhi + Ahi*Blo + Alo*Bhi  (3 WMMAs / tile).
__global__ __launch_bounds__(BLOCK_THREADS)
void gemm_split_bf16(const __bf16* __restrict__ Ahi, const __bf16* __restrict__ Alo,
                     const __bf16* __restrict__ Whi, const __bf16* __restrict__ Wlo,
                     const float*  __restrict__ bias,
                     float* __restrict__ U) {
    const int lane  = threadIdx.x & 31;
    const int wave  = threadIdx.x >> 5;
    const int lr    = lane & 15;
    const int k0    = (lane >> 4) * 8;     // K base for this lane half
    const int mbase = blockIdx.x * M_TILE;
    const int nwave = wave * 64;

    v8f acc[4][4] = {};                    // zero-init 16 accumulators

    for (int kk = 0; kk < DDIM; kk += 32) {
        v16bf Ah[4], Al[4];
#pragma unroll
        for (int ms = 0; ms < 4; ++ms) {
            const size_t ro = (size_t)(mbase + ms * 16 + lr) * DDIM + kk;
            Ah[ms] = load_frag(Ahi + ro, k0);
            Al[ms] = load_frag(Alo + ro, k0);
        }
#pragma unroll
        for (int ns = 0; ns < 4; ++ns) {
            const size_t ro = (size_t)(nwave + ns * 16 + lr) * DDIM + kk;
            v16bf Bh = load_frag(Whi + ro, k0);
            v16bf Bl = load_frag(Wlo + ro, k0);
#pragma unroll
            for (int ms = 0; ms < 4; ++ms) {
                acc[ms][ns] = __builtin_amdgcn_wmma_f32_16x16x32_bf16(
                    false, Ah[ms], false, Bh, (short)0, acc[ms][ns], false, false);
                acc[ms][ns] = __builtin_amdgcn_wmma_f32_16x16x32_bf16(
                    false, Ah[ms], false, Bl, (short)0, acc[ms][ns], false, false);
                acc[ms][ns] = __builtin_amdgcn_wmma_f32_16x16x32_bf16(
                    false, Al[ms], false, Bh, (short)0, acc[ms][ns], false, false);
            }
        }
    }

    // C/D layout: VGPR i -> row (i + 8*(lane>=16)), col = lane&15.
    const int rb = (lane >> 4) * 8;
#pragma unroll
    for (int ms = 0; ms < 4; ++ms) {
#pragma unroll
        for (int ns = 0; ns < 4; ++ns) {
            const int   n  = nwave + ns * 16 + lr;
            const float bv = bias[n];
#pragma unroll
            for (int i = 0; i < 8; ++i) {
                const int m = mbase + ms * 16 + rb + i;
                U[(size_t)m * DDIM + n] = acc[ms][ns][i] + bv;
            }
        }
    }
}

// ------------------------- elementwise recurrence (scan over T) ------------
// h_t = relu(u_t + w * h_{t-1}); one thread per 4 channels, in-place capable.
// Next step's u is prefetched so the carried dep is only fma+max.
__global__ __launch_bounds__(256)
void indrnn_scan(const float* U, const float* __restrict__ w_hh, float* Hout) {
    const int idx    = blockIdx.x * blockDim.x + threadIdx.x;    // 0..4095
    const int stride = BATCH * DDIM / 4;                         // 4096
    if (idx >= stride) return;
    const int   hq = idx & (DDIM / 4 - 1);
    const float4 w = ((const float4*)w_hh)[hq];
    const float4* Uv = (const float4*)U;
    float4*       Hv = (float4*)Hout;

    float4 h  = make_float4(0.f, 0.f, 0.f, 0.f);
    float4 un = Uv[idx];
    for (int t = 0; t < T_SEQ; ++t) {
        float4 u = un;
        if (t + 1 < T_SEQ) un = Uv[(size_t)(t + 1) * stride + idx];  // prefetch
        h.x = fmaxf(0.f, fmaf(w.x, h.x, u.x));
        h.y = fmaxf(0.f, fmaf(w.y, h.y, u.y));
        h.z = fmaxf(0.f, fmaf(w.z, h.z, u.z));
        h.w = fmaxf(0.f, fmaf(w.w, h.w, u.w));
        Hv[(size_t)t * stride + idx] = h;
    }
}

// ---------------------------------------------------------------------------
extern "C" void kernel_launch(void* const* d_in, const int* in_sizes, int n_in,
                              void* d_out, int out_size, void* d_ws, size_t ws_size,
                              hipStream_t stream) {
    const float* x     = (const float*)d_in[0];
    const float* w_ih0 = (const float*)d_in[1];
    const float* w_hh0 = (const float*)d_in[2];
    const float* b_ih0 = (const float*)d_in[3];
    const float* w_ih1 = (const float*)d_in[4];
    const float* w_hh1 = (const float*)d_in[5];
    const float* b_ih1 = (const float*)d_in[6];
    float* out = (float*)d_out;

    const size_t N_X = (size_t)M_TOTAL * DDIM;   // 33,554,432 activations
    const size_t N_W = (size_t)DDIM * DDIM;      //    262,144 weights

    // Workspace layout (~270 MB): U f32 | AHi | ALo | WH0 | WL0 | WH1 | WL1
    float*  U   = (float*)d_ws;
    __bf16* AHi = (__bf16*)((char*)d_ws + N_X * sizeof(float));
    __bf16* ALo = AHi + N_X;
    __bf16* WH0 = ALo + N_X;
    __bf16* WL0 = WH0 + N_W;
    __bf16* WH1 = WL0 + N_W;
    __bf16* WL1 = WH1 + N_W;

    const int x4 = (int)(N_X / 4), w4 = (int)(N_W / 4);
    dim3 gX((x4 + 255) / 256), gW((w4 + 255) / 256), b256(256);
    dim3 gGemm(M_TOTAL / M_TILE), bGemm(BLOCK_THREADS);   // 1024 blocks
    dim3 gScan(BATCH * DDIM / 4 / 256);                   // 16 blocks

    // Layer 0
    split_kernel<<<gX, b256, 0, stream>>>(x, AHi, ALo, x4);
    split_kernel<<<gW, b256, 0, stream>>>(w_ih0, WH0, WL0, w4);
    split_kernel<<<gW, b256, 0, stream>>>(w_ih1, WH1, WL1, w4);
    gemm_split_bf16<<<gGemm, bGemm, 0, stream>>>(AHi, ALo, WH0, WL0, b_ih0, U);
    indrnn_scan<<<gScan, b256, 0, stream>>>(U, w_hh0, U);            // h1 in U

    // Layer 1 (reuse hi/lo planes for h1)
    split_kernel<<<gX, b256, 0, stream>>>(U, AHi, ALo, x4);
    gemm_split_bf16<<<gGemm, bGemm, 0, stream>>>(AHi, ALo, WH1, WL1, b_ih1, out);
    indrnn_scan<<<gScan, b256, 0, stream>>>(out, w_hh1, out);        // in place
}
